// TransformerPoseNet_5025111736818
// MI455X (gfx1250) — matrix operand order repaired and checked by reading, not compiled, not run on hardware
//
#include <hip/hip_runtime.h>
#include <math.h>

// ---------------------------------------------------------------------------
// Fused DiT-PoseNet forward for MI455X (gfx1250, wave32, WMMA).
// N=1 sequence => attention == V-projection; whole net is a per-row MLP.
// ~1.15 TFLOP, ~260MB mandatory HBM traffic => compute / L2 bound.
//
// This revision removes the two instruction-mix bottlenecks seen in round 1:
//  * weights are pre-packed (once per launch, into d_ws ~17MB) to f16 in
//    WMMA-fragment-native layout: a B fragment = one contiguous 32B load per
//    lane (coalesced 1KB per wave), zero converts in the hot loop.
//  * erf-gelu replaced by branch-free tanh-gelu (err ~3e-4 << f16 eps).
//  * A fragments = two ds_load_b128 (ISA layout is 2x8 contiguous halfs).
//  * sc/shift modulation chunks fused to share A fragments.
// ---------------------------------------------------------------------------

typedef _Float16 f16;
typedef _Float16 v8h  __attribute__((ext_vector_type(8)));
typedef _Float16 v16h __attribute__((ext_vector_type(16)));
typedef float    v8f  __attribute__((ext_vector_type(8)));

#define HID     256
#define PDIM    1024
#define ADIM    7
#define MLPH    1024
#define DEPTH   8
#define ROWS    16              // rows per wave (== WMMA M)
#define WAVES   2               // waves per block
#define ROWSPB  (ROWS*WAVES)

struct BlockP {
  const f16 *adaP, *qkvP, *projP, *fc1P, *fc2P;          // packed f16 weights
  const float *ada_b, *qkv_b, *proj_b, *fc1_b, *fc2_b;   // f32 biases
};
struct Params {
  const float *g, *c, *mask, *t;
  BlockP blk[DEPTH];
  const f16 *pn1P, *pn2P, *up2P, *dn1P;
  const float *pn1_b, *pn2_b, *up1_w, *up1_b, *up2_b, *dn1_b, *dn2_w, *dn2_b;
  float *out;
};

// ---------------- weight packing -------------------------------------------
// Packed layout: fragment f = (nt*ktiles + kt); lane l; element j(0..15):
//   dst[(f*32 + l)*16 + j] = W[kt*32 + (j<8 ? (l>>4)*8 + j : 16 + (l>>4)*8 + j-8)][nt*16 + (l&15)]
// => B fragment load = *((const v16h*)Wp + f*32 + lane)  (32B, 32B-aligned)

__global__ void pack_w(const float* __restrict__ src, f16* __restrict__ dst,
                       int ktiles, int ld) {
  int tile = blockIdx.x;              // nt*ktiles + kt
  int nt = tile / ktiles, kt = tile % ktiles;
  int tid = threadIdx.x;              // 512 threads: j fastest for coalescing
  int j = tid & 15, lane = tid >> 4;
  int row = kt * 32 + (j < 8 ? (lane >> 4) * 8 + j : 16 + (lane >> 4) * 8 + (j - 8));
  int col = nt * 16 + (lane & 15);
  dst[(size_t)tile * 512 + tid] = (f16)src[(size_t)row * ld + col];
}

// ---------------- WMMA helpers (layouts per CDNA5 ISA) ----------

__device__ __forceinline__ v8f zero8() {
  v8f z = {0.f,0.f,0.f,0.f,0.f,0.f,0.f,0.f};
  return z;
}
__device__ __forceinline__ v8f wmma_f16(v16h a, v16h b, v8f c) {
  return __builtin_amdgcn_wmma_f32_16x16x32_f16(false, a, false, b,
                                                (short)0, c, false, false);
}

// 16-bit A 16x32: per lane two contiguous 8-half runs -> 2x ds_load_b128.
__device__ __forceinline__ v16h frag_a_lds(const f16* A, int lda, int k0, int lane) {
  int m = lane & 15, h = lane >> 4;
  const f16* base = A + m * lda + k0 + h * 8;
  v8h lo = *(const v8h*)base;
  v8h hi = *(const v8h*)(base + 16);
  return __builtin_shufflevector(lo, hi, 0,1,2,3,4,5,6,7,8,9,10,11,12,13,14,15);
}

// acc += A(16 x 32*ktn from LDS, local k origin 0) * Wp[nt, kt0..kt0+ktn)
__device__ __forceinline__ v8f gemm_pk(v8f acc, const f16* A, int lda,
                                       const f16* Wp, int ktiles, int nt,
                                       int kt0, int ktn, int lane) {
  const v16h* p = (const v16h*)Wp + ((size_t)nt * ktiles + kt0) * 32 + lane;
  for (int kt = 0; kt < ktn; ++kt) {
    acc = wmma_f16(frag_a_lds(A, lda, kt * 32, lane), p[0], acc);
    p += 32;
  }
  return acc;
}

// ---------------- activations (branch-free) --------------------------------

__device__ __forceinline__ float tanh_fast(float x) {
  float xc = fminf(fmaxf(x, -15.f), 15.f);
  float e = __expf(2.0f * xc);
  return (e - 1.0f) / (e + 1.0f);
}
__device__ __forceinline__ float gelu_f(float x) {      // tanh form; used for
  return 0.5f * x * (1.0f + tanh_fast(0.7978845608028654f *  // both exact and
                     (x + 0.044715f * x * x * x)));          // approx sites
}
__device__ __forceinline__ float silu_f(float x) {
  return x / (1.0f + __expf(-x));
}

// LayerNorm(256) for 16 rows: 2 lanes/row, partner reduce via shfl_xor(16).
__device__ __forceinline__ void layernorm16(const f16* X, f16* H, int lane) {
  int row = lane & 15, h = lane >> 4;
  const f16* p = X + row * HID + h * 128;
  float s = 0.f, ss = 0.f;
#pragma unroll 8
  for (int j = 0; j < 128; j++) { float v = (float)p[j]; s += v; ss += v * v; }
  s  += __shfl_xor(s, 16, 32);
  ss += __shfl_xor(ss, 16, 32);
  float mean = s * (1.0f / 256.0f);
  float var  = ss * (1.0f / 256.0f) - mean * mean;
  float rstd = rsqrtf(var + 1e-6f);
  f16* q = H + row * HID + h * 128;
#pragma unroll 8
  for (int j = 0; j < 128; j++) q[j] = (f16)(((float)p[j] - mean) * rstd);
}

// Fused modulation: H = H*(1 + sc) + s   (sc at tile col ntScale, s at ntShift)
// Shares A fragments between the two ada GEMM chunks.
__device__ __forceinline__ void apply_mod2(const f16* sC, const f16* adaP,
                                           const float* ada_b, int ntScale,
                                           int ntShift, f16* H, int lane) {
  int n15 = lane & 15, mb = (lane >> 4) * 8;
  for (int t = 0; t < 16; t++) {
    const v16h* pS = (const v16h*)adaP + (size_t)(ntScale + t) * 8 * 32 + lane;
    const v16h* pH = (const v16h*)adaP + (size_t)(ntShift + t) * 8 * 32 + lane;
    v8f aS = zero8(), aH = zero8();
    for (int kt = 0; kt < 8; kt++) {
      v16h a = frag_a_lds(sC, HID, kt * 32, lane);
      aS = wmma_f16(a, pS[(size_t)kt * 32], aS);
      aH = wmma_f16(a, pH[(size_t)kt * 32], aH);
    }
    int n = n15 + t * 16;
    float bS = ada_b[ntScale * 16 + n];
    float bH = ada_b[ntShift * 16 + n];
#pragma unroll
    for (int r = 0; r < 8; r++) {
      int off = (mb + r) * HID + n;
      float hv = (float)H[off];
      H[off] = (f16)(hv * (1.0f + aS[r] + bS) + aH[r] + bH);
    }
  }
}

// ---------------- main kernel ----------------------------------------------

__global__ __launch_bounds__(WAVES * 32)
void posenet_fused(Params P) {
  // per-wave LDS: xA(16x256) hA(16x256) sC(16x256) mC(16x256) f16 = 32KB
  __shared__ __align__(16) f16 lds[WAVES * 16384];
  int lane = threadIdx.x & 31;
  int wave = threadIdx.x >> 5;
  f16* xA = lds + wave * 16384;
  f16* hA = xA + 4096;
  f16* sC = xA + 8192;
  f16* mC = xA + 12288;
  int r0 = (blockIdx.x * WAVES + wave) * ROWS;
  int n15 = lane & 15, mb = (lane >> 4) * 8;

  // ---- pn1 (1024->256, gelu): stage masked c in LDS (two 16x512 halves) ----
  {
    v8f accT[16];
#pragma unroll
    for (int t = 0; t < 16; t++) accT[t] = zero8();
    f16* buf = sC;                       // sC+mC region = 8192 halfs = 16x512
    for (int kh = 0; kh < 2; kh++) {
      for (int idx = lane * 4; idx < 16 * 512; idx += 128) {
        int m = idx >> 9, k = idx & 511;
        float4 v = *(const float4*)(P.c + (size_t)(r0 + m) * PDIM + kh * 512 + k);
        float mv = P.mask[r0 + m];
        buf[m * 512 + k    ] = (f16)(v.x * mv);
        buf[m * 512 + k + 1] = (f16)(v.y * mv);
        buf[m * 512 + k + 2] = (f16)(v.z * mv);
        buf[m * 512 + k + 3] = (f16)(v.w * mv);
      }
      for (int t = 0; t < 16; t++)
        accT[t] = gemm_pk(accT[t], buf, 512, P.pn1P, 32, t, kh * 16, 16, lane);
    }
    for (int t = 0; t < 16; t++) {
      int n = n15 + t * 16;
      float b = P.pn1_b[n];
#pragma unroll
      for (int r = 0; r < 8; r++)
        hA[(mb + r) * HID + n] = (f16)gelu_f(accT[t][r] + b);
    }
  }
  // ---- pn2 (256->256) + time embedding + silu -> sC ----
  {
    const float lfac = -9.210340371976184f / 127.0f;  // -ln(10000)/(half-1)
    for (int t = 0; t < 16; t++) {
      v8f acc = gemm_pk(zero8(), hA, HID, P.pn2P, 8, t, 0, 8, lane);
      int n = n15 + t * 16;
      float b = P.pn2_b[n];
      int j = (n < 128) ? n : n - 128;
      float fr = __expf(lfac * (float)j);
#pragma unroll
      for (int r = 0; r < 8; r++) {
        float ang = P.t[r0 + mb + r] * fr;
        float emb = (n < 128) ? __sinf(ang) : __cosf(ang);
        sC[(mb + r) * HID + n] = (f16)silu_f(acc[r] + b + emb);
      }
    }
  }
  // ---- up1 (7->128, gelu) scalar -> mC ----
  for (int idx = lane; idx < 16 * 128; idx += 32) {
    int m = idx >> 7, n = idx & 127;
    float s = P.up1_b[n];
#pragma unroll
    for (int k = 0; k < ADIM; k++)
      s += P.g[(size_t)(r0 + m) * ADIM + k] * P.up1_w[k * 128 + n];
    mC[m * 128 + n] = (f16)gelu_f(s);
  }
  // ---- up2 (128->256) -> xA ----
  for (int t = 0; t < 16; t++) {
    v8f acc = gemm_pk(zero8(), mC, 128, P.up2P, 4, t, 0, 4, lane);
    int n = n15 + t * 16;
    float b = P.up2_b[n];
#pragma unroll
    for (int r = 0; r < 8; r++)
      xA[(mb + r) * HID + n] = (f16)(acc[r] + b);
  }

  // ---- transformer blocks (N=1 => attn == V then proj) ----
  // ada col tiles: s1=0, sc1=16, g1=32, s2=48, sc2=64, g2=80
  for (int d = 0; d < DEPTH; d++) {
    BlockP bp = P.blk[d];

    layernorm16(xA, hA, lane);
    apply_mod2(sC, bp.adaP, bp.ada_b, 16, 0, hA, lane);    // h = ln*(1+sc1)+s1

    // v = h @ qkv_v + b -> mC
    for (int t = 0; t < 16; t++) {
      v8f acc = gemm_pk(zero8(), hA, HID, bp.qkvP, 8, t, 0, 8, lane);
      int n = n15 + t * 16;
      float b = bp.qkv_b[2 * HID + n];
#pragma unroll
      for (int r = 0; r < 8; r++)
        mC[(mb + r) * HID + n] = (f16)(acc[r] + b);
    }
    // g1 -> hA (h dead)
    for (int t = 0; t < 16; t++) {
      v8f acc = gemm_pk(zero8(), sC, HID, bp.adaP, 8, 32 + t, 0, 8, lane);
      int n = n15 + t * 16;
      float b = bp.ada_b[512 + n];
#pragma unroll
      for (int r = 0; r < 8; r++)
        hA[(mb + r) * HID + n] = (f16)(acc[r] + b);
    }
    // x += g1 * (v @ proj + b)
    for (int t = 0; t < 16; t++) {
      v8f acc = gemm_pk(zero8(), mC, HID, bp.projP, 8, t, 0, 8, lane);
      int n = n15 + t * 16;
      float b = bp.proj_b[n];
#pragma unroll
      for (int r = 0; r < 8; r++) {
        int off = (mb + r) * HID + n;
        xA[off] = (f16)((float)xA[off] + (float)hA[off] * (acc[r] + b));
      }
    }

    layernorm16(xA, hA, lane);
    apply_mod2(sC, bp.adaP, bp.ada_b, 64, 48, hA, lane);   // h = ln*(1+sc2)+s2

    // MLP: hidden chunked by 256, fc2 output accumulated in registers.
    v8f out_acc[16];
#pragma unroll
    for (int t = 0; t < 16; t++) out_acc[t] = zero8();
    for (int hc = 0; hc < 4; hc++) {
      for (int t = 0; t < 16; t++) {
        v8f acc = gemm_pk(zero8(), hA, HID, bp.fc1P, 8, hc * 16 + t, 0, 8, lane);
        int n = n15 + t * 16;
        float b = bp.fc1_b[hc * 256 + n];
#pragma unroll
        for (int r = 0; r < 8; r++)
          mC[(mb + r) * HID + n] = (f16)gelu_f(acc[r] + b);
      }
      for (int t = 0; t < 16; t++)
        out_acc[t] = gemm_pk(out_acc[t], mC, HID, bp.fc2P, 32, t, hc * 8, 8, lane);
    }
    // g2 -> mC
    for (int t = 0; t < 16; t++) {
      v8f acc = gemm_pk(zero8(), sC, HID, bp.adaP, 8, 80 + t, 0, 8, lane);
      int n = n15 + t * 16;
      float b = bp.ada_b[1280 + n];
#pragma unroll
      for (int r = 0; r < 8; r++)
        mC[(mb + r) * HID + n] = (f16)(acc[r] + b);
    }
    // x += g2 * (out + fc2_b)
    for (int t = 0; t < 16; t++) {
      int n = n15 + t * 16;
      float b = bp.fc2_b[n];
#pragma unroll
      for (int r = 0; r < 8; r++) {
        int off = (mb + r) * HID + n;
        xA[off] = (f16)((float)xA[off] + (float)mC[off] * (out_acc[t][r] + b));
      }
    }
  }

  // ---- head: dn1 (256->128, gelu) -> hA[16x128] ----
  for (int t = 0; t < 8; t++) {
    v8f acc = gemm_pk(zero8(), xA, HID, P.dn1P, 8, t, 0, 8, lane);
    int n = n15 + t * 16;
    float b = P.dn1_b[n];
#pragma unroll
    for (int r = 0; r < 8; r++)
      hA[(mb + r) * 128 + n] = (f16)gelu_f(acc[r] + b);
  }
  // ---- dn2 (128->7) scalar, f32 output ----
  for (int idx = lane; idx < 16 * ADIM; idx += 32) {
    int m = idx / ADIM, n = idx % ADIM;
    float s = P.dn2_b[n];
#pragma unroll 8
    for (int k = 0; k < 128; k++)
      s += (float)hA[m * 128 + k] * P.dn2_w[k * ADIM + n];
    P.out[(size_t)(r0 + m) * ADIM + n] = s;
  }
}

// ---------------- launch ----------------------------------------------------
// Input order: top-level dict insertion order (g, c, context_mask, _t, params),
// params as a jax pytree: 'blocks' first (8 dicts, keys alphabetical:
// ada_b, ada_w, fc1_b, fc1_w, fc2_b, fc2_w, proj_b, proj_w, qkv_b, qkv_w),
// then dn1_b, dn1_w, dn2_b, dn2_w, pn1_b, pn1_w, pn2_b, pn2_w,
// up1_b, up1_w, up2_b, up2_w.  (96 arrays total)
// d_ws usage: ~17.6MB of packed f16 weights (re-packed every call).

extern "C" void kernel_launch(void* const* d_in, const int* in_sizes, int n_in,
                              void* d_out, int out_size, void* d_ws, size_t ws_size,
                              hipStream_t stream) {
  Params P;
  P.g    = (const float*)d_in[0];
  P.c    = (const float*)d_in[1];
  P.mask = (const float*)d_in[2];
  P.t    = (const float*)d_in[3];

  f16* ws = (f16*)d_ws;
  size_t off = 0;
  auto pack = [&](const float* src, int K, int N, int ld) -> const f16* {
    f16* dst = ws + off;
    off += (size_t)K * N;
    int kt = K / 32, nt = N / 16;
    hipLaunchKernelGGL(pack_w, dim3(kt * nt), dim3(512), 0, stream, src, dst, kt, ld);
    return dst;
  };

  int i = 4;
  const float* raw[DEPTH][10];
  for (int d = 0; d < DEPTH; d++)
    for (int k = 0; k < 10; k++) raw[d][k] = (const float*)d_in[i++];
  // order: ada_b, ada_w, fc1_b, fc1_w, fc2_b, fc2_w, proj_b, proj_w, qkv_b, qkv_w
  for (int d = 0; d < DEPTH; d++) {
    P.blk[d].ada_b  = raw[d][0];
    P.blk[d].adaP   = pack(raw[d][1], HID, 6 * HID, 6 * HID);
    P.blk[d].fc1_b  = raw[d][2];
    P.blk[d].fc1P   = pack(raw[d][3], HID, MLPH, MLPH);
    P.blk[d].fc2_b  = raw[d][4];
    P.blk[d].fc2P   = pack(raw[d][5], MLPH, HID, HID);
    P.blk[d].proj_b = raw[d][6];
    P.blk[d].projP  = pack(raw[d][7], HID, HID, HID);
    P.blk[d].qkv_b  = raw[d][8];
    P.blk[d].qkvP   = pack(raw[d][9] + 2 * HID, HID, HID, 3 * HID); // V slice
  }
  P.dn1_b = (const float*)d_in[i++];
  P.dn1P  = pack((const float*)d_in[i++], HID, 128, 128);
  P.dn2_b = (const float*)d_in[i++];
  P.dn2_w = (const float*)d_in[i++];
  P.pn1_b = (const float*)d_in[i++];
  P.pn1P  = pack((const float*)d_in[i++], PDIM, HID, HID);
  P.pn2_b = (const float*)d_in[i++];
  P.pn2P  = pack((const float*)d_in[i++], HID, HID, HID);
  P.up1_b = (const float*)d_in[i++];
  P.up1_w = (const float*)d_in[i++];
  P.up2_b = (const float*)d_in[i++];
  P.up2P  = pack((const float*)d_in[i++], 128, HID, HID);
  P.out = (float*)d_out;

  int rows = in_sizes[0] / ADIM;            // 65536
  dim3 grid(rows / ROWSPB), block(WAVES * 32);
  hipLaunchKernelGGL(posenet_fused, grid, block, 0, stream, P);
}
